// AdaPKC2D_Thre_34316788695832
// MI455X (gfx1250) — matrix operand order repaired and checked by reading, not compiled, not run on hardware
//
#include <hip/hip_runtime.h>
#include <hip/hip_bf16.h>

#define BB    2
#define INCH  16
#define OUTCH 32
#define HH    256
#define WW    256
#define NOFF  16
#define NGRID 16
#define PLANE (HH*WW)          // 65536 elements per channel plane

typedef __attribute__((ext_vector_type(16))) __bf16 v16bf;
typedef __attribute__((ext_vector_type(8)))  float  v8f;

// ---------------------------------------------------------------------------
// Compile-time replication of _uni_prf_grid / _build_offsets (rb=1).
// idx5 tables are np.round(linspace(0, 2*r, 5)) with banker's rounding:
//   r=2: [0,1,2,3,4]  r=3: [0,2,3,4,6]  r=4: [0,2,4,6,8]  r=5: [0,2,5,8,10]
// Ring order: top row (5), right col (3), bottom row (5), left col (3) = 16.
// INIT_INDEX (gh=gw=1) is grid 0.
// ---------------------------------------------------------------------------
struct Offsets { int dx[NGRID][NOFF]; int dy[NGRID][NOFF]; };

static constexpr int IDX5[4][5] = {
  {0, 1, 2, 3, 4},    // half-extent 2
  {0, 2, 3, 4, 6},    // half-extent 3  (1.5->2, 4.5->4 banker's)
  {0, 2, 4, 6, 8},    // half-extent 4
  {0, 2, 5, 8, 10}    // half-extent 5  (2.5->2, 7.5->8 banker's)
};

constexpr Offsets make_offsets() {
  Offsets o{};
  for (int gi = 0; gi < 4; ++gi) {
    for (int gj = 0; gj < 4; ++gj) {
      const int g = gi * 4 + gj;
      const int r = gi + 2;              // rb + gh, gh = gi+1
      const int c = gj + 2;              // rb + gw
      const int* td = IDX5[c - 2];       // column picks (5)
      const int* lr = IDX5[r - 2];       // row picks, interior [1..3]
      int n = 0;
      for (int j = 0; j < 5; ++j) { o.dx[g][n] = -r;        o.dy[g][n] = td[j] - c; ++n; }
      for (int i = 1; i < 4; ++i) { o.dx[g][n] = lr[i] - r; o.dy[g][n] =  c;        ++n; }
      for (int j = 0; j < 5; ++j) { o.dx[g][n] =  r;        o.dy[g][n] = td[j] - c; ++n; }
      for (int i = 1; i < 4; ++i) { o.dx[g][n] = lr[i] - r; o.dy[g][n] = -c;        ++n; }
    }
  }
  return o;
}

__constant__ Offsets OFFS = make_offsets();

__device__ __forceinline__ int clampi(int v, int lo, int hi) {
  return v < lo ? lo : (v > hi ? hi : v);
}

// ---------------------------------------------------------------------------
// Kernel 1: per-pixel guard-band selection (sims -> stable sort -> max gap).
// Branch-free gathers: clamp coordinate, load unconditionally, scale dot by
// an in-bounds mask (OOB => dot=0 => sigmoid(0)=0.5, matching zero padding).
// ---------------------------------------------------------------------------
__global__ void adapkc_select_kernel(const float* __restrict__ x,
                                     int* __restrict__ ori) {
  const int p = blockIdx.x * blockDim.x + threadIdx.x;
  if (p >= BB * HH * WW) return;
  const int b  = p >> 16;          // H*W = 65536
  const int hw = p & 0xFFFF;
  const int h  = hw >> 8;
  const int w  = hw & 0xFF;

  const float* xb = x + (size_t)b * INCH * PLANE;
  float xc[INCH];
#pragma unroll
  for (int cc = 0; cc < INCH; ++cc) xc[cc] = xb[cc * PLANE + h * WW + w];

  float sim[NGRID];
  for (int g = 0; g < NGRID; ++g) {
    float s = 0.f;
#pragma unroll
    for (int n = 0; n < NOFF; ++n) {
      const int hh = h + OFFS.dx[g][n];
      const int ww = w + OFFS.dy[g][n];
      const bool inb = ((unsigned)hh < HH) & ((unsigned)ww < WW);
      const float* bp = xb + clampi(hh, 0, HH - 1) * WW + clampi(ww, 0, WW - 1);
      float dot = 0.f;
#pragma unroll
      for (int cc = 0; cc < INCH; ++cc)        // all-immediate plane offsets
        dot += bp[cc * PLANE] * xc[cc];
      dot *= inb ? (1.0f / INCH) : 0.0f;       // OOB ring point contributes 0
      s += 1.0f / (1.0f + __expf(-dot));       // sigmoid(0)=0.5 matches pad
    }
    sim[g] = s * (1.0f / NOFF);
  }

  // stable ascending insertion sort with index tracking (== jnp.argsort)
  float sv[NGRID]; int so[NGRID];
#pragma unroll
  for (int i = 0; i < NGRID; ++i) { sv[i] = sim[i]; so[i] = i; }
  for (int i = 1; i < NGRID; ++i) {
    const float kv = sv[i]; const int ko = so[i];
    int j = i - 1;
    while (j >= 0 && sv[j] > kv) { sv[j + 1] = sv[j]; so[j + 1] = so[j]; --j; }
    sv[j + 1] = kv; so[j + 1] = ko;
  }
  float bestd = sv[1] - sv[0]; int besti = 0;
  for (int j = 1; j < NGRID - 1; ++j) {
    const float d = sv[j + 1] - sv[j];
    if (d > bestd) { bestd = d; besti = j; }   // first max, like argmax
  }
  ori[p] = (bestd >= 0.001f) ? so[besti] : 0;  // INIT_INDEX = 0
}

// ---------------------------------------------------------------------------
// Kernel 2: out[b,o,h,w] = bias[o] + sum_{c,n} (x_c - x_r[sel]) * w[o,c,n]
// One wave32 per 16-pixel tile. WMMA f32 <- bf16 16x16x32, K=256 in 8 steps,
// two M-tiles (OUT_CH=32). A/B hold K = 16*half + s (contiguous, not the ISA table order); D per ISA.
// Neighbor pointers (with this lane's K-half channel plane folded in) and
// float masks are precomputed once; all K-loop loads then use compile-time
// immediate offsets (kb*2*PLANE*4 <= 3.67MB < 24-bit IOFFSET range).
// ---------------------------------------------------------------------------
__global__ void adapkc_wmma_kernel(const float* __restrict__ x,
                                   const float* __restrict__ weight,
                                   const float* __restrict__ bias,
                                   const int* __restrict__ ori,
                                   float* __restrict__ out) {
  const int wave  = blockIdx.x * (blockDim.x >> 5) + (threadIdx.x >> 5);
  const int lane  = threadIdx.x & 31;
  const int col   = lane & 15;     // pixel column within tile / D column
  const int khalf = lane >> 4;     // which K-half this lane holds

  // tile -> (b, h, w0): 4096 tiles per image, 16 tiles per row
  const int b  = wave >> 12;
  const int tr = wave & 4095;
  const int h  = tr >> 4;
  const int w0 = (tr & 15) << 4;
  const int w  = w0 + col;         // this lane's pixel

  // base plane for this lane: batch + K-half channel offset folded in
  const float* xh = x + (size_t)b * (INCH * PLANE) + khalf * PLANE;

  const int p = (b << 16) | (h << 8) | w;
  const int g = ori[p];

  // branch-free gather setup: clamped pointers + masks, computed once
  const float* xpix[NOFF];
  float fmsk[NOFF];
#pragma unroll
  for (int n = 0; n < NOFF; ++n) {
    const int hh = h + OFFS.dx[g][n];
    const int ww = w + OFFS.dy[g][n];
    const bool inb = ((unsigned)hh < HH) & ((unsigned)ww < WW);
    xpix[n] = xh + clampi(hh, 0, HH - 1) * WW + clampi(ww, 0, WW - 1);
    fmsk[n] = inb ? 1.0f : 0.0f;
  }
  const float* xcen = xh + h * WW + w;

  // weight row base for this lane (K-half folded in); rest is immediates
  const int orow = lane & 15;
  const float* wr0 = weight + (orow)      * 256 + khalf * 16;
  const float* wr1 = weight + (orow + 16) * 256 + khalf * 16;

  v8f acc0 = {};   // outputs 0..15
  v8f acc1 = {};   // outputs 16..31

#pragma unroll
  for (int kb = 0; kb < 8; ++kb) {
    const int poff = kb * 2 * PLANE;           // compile-time per iteration

    // ---- B fragment: x_prf, 32(K) x 16(pixels) bf16.
    // Lane half selects K 0..15 vs 16..31; each half-block is one channel
    // (c = 2*kb + khalf) across the 16 ring offsets n.
    const float xcv = xcen[poff];
    v16bf bfrag;
#pragma unroll
    for (int n = 0; n < NOFF; ++n) {
      const float xr = xpix[n][poff] * fmsk[n];   // unconditional load + mask
      bfrag[n] = (__bf16)(xcv - xr);
    }

    // ---- A fragments: weight rows, 16(M) x 32(K) bf16.
    // element 2v+t holds K = kb*32 + 2v+t + khalf*16.
    v16bf a0, a1;
#pragma unroll
    for (int v = 0; v < 8; ++v) {
#pragma unroll
      for (int t = 0; t < 2; ++t) {
        const int k = kb * 32 + 2 * v + t; // immediate
        a0[2 * v + t] = (__bf16)wr0[k];
        a1[2 * v + t] = (__bf16)wr1[k];
      }
    }

    acc0 = __builtin_amdgcn_wmma_f32_16x16x32_bf16(false, a0, false, bfrag,
                                                   (short)0, acc0, false, false);
    acc1 = __builtin_amdgcn_wmma_f32_16x16x32_bf16(false, a1, false, bfrag,
                                                   (short)0, acc1, false, false);
  }

  // ---- D layout: VGPR r, lane half -> row r + 8*khalf; column = lane&15.
#pragma unroll
  for (int r = 0; r < 8; ++r) {
    const int m  = r + khalf * 8;
    const int o0 = m;
    const int o1 = m + 16;
    out[(((size_t)b * OUTCH + o0) * HH + h) * WW + w] = acc0[r] + bias[o0];
    out[(((size_t)b * OUTCH + o1) * HH + h) * WW + w] = acc1[r] + bias[o1];
  }
}

// ---------------------------------------------------------------------------
extern "C" void kernel_launch(void* const* d_in, const int* in_sizes, int n_in,
                              void* d_out, int out_size, void* d_ws, size_t ws_size,
                              hipStream_t stream) {
  const float* x      = (const float*)d_in[0];
  const float* weight = (const float*)d_in[1];
  const float* bias   = (const float*)d_in[2];
  float* out = (float*)d_out;
  int*   ori = (int*)d_ws;                       // B*H*W ints = 512 KB

  const int npix = BB * HH * WW;
  adapkc_select_kernel<<<(npix + 255) / 256, 256, 0, stream>>>(x, ori);

  // 8192 wave-tiles, 8 waves (256 threads) per block
  adapkc_wmma_kernel<<<8192 / 8, 256, 0, stream>>>(x, weight, bias, ori, out);
}
